// GNNModel_66623532696338
// MI455X (gfx1250) — compile-verified, matrix operand order checked by reading, not run on hardware
//
#include <hip/hip_runtime.h>
#include <hip/hip_bf16.h>
#include <math.h>

#define EPS_F 1e-6f

typedef __attribute__((ext_vector_type(16))) _Float16 v16h;
typedef __attribute__((ext_vector_type(8)))  float    v8f;

// ---------------------------------------------------------------------------
// Load a 16x32 f16 A/B fragment from an fp32 row-major matrix row.
// ISA layout (CDNA5 ISA, 16-bit A 16x32):
//   lane = M (or N for B);  half = lane>>4 selects the K sub-pattern:
//   v<4 : K = half*8 + 2v .. +1      v>=4 : K = 16 + half*8 + 2(v-4) .. +1
// ---------------------------------------------------------------------------
__device__ __forceinline__ v16h load_frag_f16(const float* __restrict__ rowbase,
                                              int k0, int half) {
  v16h a;
#pragma unroll
  for (int v = 0; v < 8; ++v) {
    const int kb = k0 + ((v < 4) ? (half * 8 + v * 2)
                                 : (16 + half * 8 + (v - 4) * 2));
    const float2 f = *(const float2*)(rowbase + kb);
    a[2 * v]     = (_Float16)f.x;
    a[2 * v + 1] = (_Float16)f.y;
  }
  return a;
}

// ---------------------------------------------------------------------------
// One-time weight prep: convert W (128x128 f32, row-major [out][in]) into
// fragment-ordered f16: WF[(tn*4 + ks)*32 + lane][16 halves].
// Launch: 1 block x 1024 threads (32 waves = 8 n-tiles x 4 k-steps).
// ---------------------------------------------------------------------------
__global__ void wprep(const float* __restrict__ W, _Float16* __restrict__ WF) {
  const int wv   = threadIdx.x >> 5;      // 0..31
  const int lane = threadIdx.x & 31;
  const int half = lane >> 4;
  const int r    = lane & 15;
  const int tn = wv >> 2;
  const int ks = wv & 3;
  v16h b = load_frag_f16(W + ((size_t)tn * 16 + r) * 128, ks * 32, half);
  *(v16h*)(WF + ((size_t)(tn * 4 + ks) * 32 + lane) * 16) = b;
}

// ---------------------------------------------------------------------------
// H[M x 128] = X[M x 128] @ W^T + bias, weights pre-packed as f16 fragments.
// One wave per 16x128 output strip: A fragment loaded once per K-step and
// reused across all 8 N-tiles (32 v_wmma per wave, 8 v8f accumulators).
// grid = ceil(Mtiles/4), block = 128 (4 waves = 4 consecutive M tiles).
// ---------------------------------------------------------------------------
__global__ void gemm128_wmma(const float* __restrict__ X,
                             const _Float16* __restrict__ WF,
                             const float* __restrict__ bias,
                             float* __restrict__ H,
                             int Mtiles) {
  const int wave = threadIdx.x >> 5;
  const int lane = threadIdx.x & 31;
  const int half = lane >> 4;
  const int r    = lane & 15;
  const int tile_m = blockIdx.x * 4 + wave;
  if (tile_m >= Mtiles) return;           // wave-uniform: EXEC stays all-1s

  const float* xrow = X + ((size_t)tile_m * 16 + r) * 128;

  v8f acc[8] = {};
#pragma unroll
  for (int ks = 0; ks < 4; ++ks) {
    v16h a = load_frag_f16(xrow, ks * 32, half);
#pragma unroll
    for (int tn = 0; tn < 8; ++tn) {
      v16h b = *(const v16h*)(WF + ((size_t)(tn * 4 + ks) * 32 + lane) * 16);
      acc[tn] = __builtin_amdgcn_wmma_f32_16x16x32_f16(false, a, false, b,
                                                       (short)0, acc[tn],
                                                       false, false);
    }
  }

  float* hrow = H + (size_t)tile_m * 16 * 128;
#pragma unroll
  for (int tn = 0; tn < 8; ++tn) {
    const float bn = bias[tn * 16 + r];
#pragma unroll
    for (int i = 0; i < 8; ++i)           // C/D layout: m = i + 8*half, n = r
      hrow[(size_t)(i + 8 * half) * 128 + tn * 16 + r] = acc[tn][i] + bn;
  }
}

// ---------------------------------------------------------------------------
// Edge pass: one wave per edge. lane handles 4 of 128 features (float4).
//   t = copysign((|h[src]|+eps)^3, h[src]);  atomicAdd(acc[dst], t)
// 25.6 MB feature table lives in L2 (192 MB), so gather+RMW stay on-chip.
// ---------------------------------------------------------------------------
__global__ void edge_pass(const float* __restrict__ H,
                          const int* __restrict__ src,
                          const int* __restrict__ dst,
                          float* __restrict__ acc,
                          float* __restrict__ cnt,
                          int E) {
  const int wave = (int)((blockIdx.x * (size_t)blockDim.x + threadIdx.x) >> 5);
  const int lane = threadIdx.x & 31;
  if (wave >= E) return;
  const int s = src[wave];
  const int d = dst[wave];

  const float4 v = ((const float4*)(H + (size_t)s * 128))[lane];
  float4 t;
  {
    float a;
    a = fabsf(v.x) + EPS_F; t.x = copysignf(a * a * a, v.x);
    a = fabsf(v.y) + EPS_F; t.y = copysignf(a * a * a, v.y);
    a = fabsf(v.z) + EPS_F; t.z = copysignf(a * a * a, v.z);
    a = fabsf(v.w) + EPS_F; t.w = copysignf(a * a * a, v.w);
  }
  float* ap = acc + (size_t)d * 128 + lane * 4;
  atomicAdd(ap + 0, t.x);
  atomicAdd(ap + 1, t.y);
  atomicAdd(ap + 2, t.z);
  atomicAdd(ap + 3, t.w);
  if (lane == 0) atomicAdd(cnt + d, 1.0f);
}

// ---------------------------------------------------------------------------
// Node finish: y = signed_pow(acc/max(cnt,1), 1/3), optional relu.
// ---------------------------------------------------------------------------
__global__ void node_finish(const float* __restrict__ acc,
                            const float* __restrict__ cnt,
                            float* __restrict__ out,
                            int total, int doRelu) {
  const int i = (int)(blockIdx.x * (size_t)blockDim.x + threadIdx.x);
  if (i >= total) return;
  const int node = i >> 7;
  const float c = fmaxf(cnt[node], 1.0f);
  const float v = acc[i] / c;
  float r = copysignf(powf(fabsf(v) + EPS_F, 1.0f / 3.0f), v);
  if (doRelu) r = fmaxf(r, 0.0f);
  out[i] = r;
}

// ---------------------------------------------------------------------------
// Node finish of conv2 fused with global gen-mean pool accumulation (p=2):
//   y = signed_pow(acc/max(cnt,1), 1/3);  t = copysign((|y|+eps)^2, y)
//   atomicAdd(gacc[batch[node]*128 + f], t); one lane counts the node.
// ---------------------------------------------------------------------------
__global__ void node_pool(const float* __restrict__ acc,
                          const float* __restrict__ cnt,
                          const int* __restrict__ batch,
                          float* __restrict__ gacc,
                          float* __restrict__ gcnt,
                          int total) {
  const int i = (int)(blockIdx.x * (size_t)blockDim.x + threadIdx.x);
  if (i >= total) return;
  const int node = i >> 7;
  const int f    = i & 127;
  const float c = fmaxf(cnt[node], 1.0f);
  const float v = acc[i] / c;
  const float y = copysignf(powf(fabsf(v) + EPS_F, 1.0f / 3.0f), v);
  const float a = fabsf(y) + EPS_F;
  const float t = copysignf(a * a, y);
  const int g = batch[node];
  atomicAdd(gacc + (size_t)g * 128 + f, t);
  if (f == 0) atomicAdd(gcnt + g, 1.0f);
}

// ---------------------------------------------------------------------------
// Per-graph head: g = signed_sqrt(gacc/max(gcnt,1)); out = g @ Wout^T + bout.
// One block (128 threads) per graph; g staged in LDS, 64 lanes do the dots.
// ---------------------------------------------------------------------------
__global__ void graph_head(const float* __restrict__ gacc,
                           const float* __restrict__ gcnt,
                           const float* __restrict__ Wout,
                           const float* __restrict__ bout,
                           float* __restrict__ out) {
  __shared__ float g[128];
  const int gi = blockIdx.x;
  const float c = fmaxf(gcnt[gi], 1.0f);
  const float v = gacc[(size_t)gi * 128 + threadIdx.x] / c;
  g[threadIdx.x] = copysignf(sqrtf(fabsf(v) + EPS_F), v);
  __syncthreads();
  if (threadIdx.x < 64) {
    const float* w = Wout + (size_t)threadIdx.x * 128;
    float s = bout[threadIdx.x];
#pragma unroll 8
    for (int k = 0; k < 128; ++k) s += g[k] * w[k];
    out[(size_t)gi * 64 + threadIdx.x] = s;
  }
}

// ---------------------------------------------------------------------------
extern "C" void kernel_launch(void* const* d_in, const int* in_sizes, int n_in,
                              void* d_out, int out_size, void* d_ws, size_t ws_size,
                              hipStream_t stream) {
  const float* x    = (const float*)d_in[0];
  const float* W1   = (const float*)d_in[1];
  const float* b1   = (const float*)d_in[2];
  const float* W2   = (const float*)d_in[3];
  const float* b2   = (const float*)d_in[4];
  const float* Wout = (const float*)d_in[5];
  const float* bout = (const float*)d_in[6];
  const int*   ei   = (const int*)d_in[7];
  const int*   bat  = (const int*)d_in[8];

  const int N = in_sizes[0] / 128;          // 50000
  const int E = in_sizes[7] / 2;            // 600000
  const int G = out_size / 64;              // 512
  const int* src = ei;
  const int* dst = ei + E;

  // workspace layout (256B aligned)
  char* ws = (char*)d_ws;
  size_t off = 0;
  auto take = [&](size_t bytes) { void* p = ws + off; off = (off + bytes + 255) & ~(size_t)255; return p; };
  float*     buf0 = (float*)take((size_t)N * 128 * sizeof(float));  // h / y ping
  float*     buf1 = (float*)take((size_t)N * 128 * sizeof(float));  // acc / h pong
  float*     cntN = (float*)take((size_t)N * sizeof(float));
  float*     gacc = (float*)take((size_t)G * 128 * sizeof(float));
  float*     gcnt = (float*)take((size_t)G * sizeof(float));
  _Float16*  WF1  = (_Float16*)take((size_t)128 * 128 * sizeof(_Float16));
  _Float16*  WF2  = (_Float16*)take((size_t)128 * 128 * sizeof(_Float16));
  (void)ws_size; (void)n_in;

  const int Mtiles  = N / 16;               // 3125
  const int gemmGrd = (Mtiles + 3) / 4;
  const int totNF   = N * 128;
  const int edgeBlk = (E + 7) / 8;          // 8 waves (edges) per 256-thr block

  // ---- weight prep (once per launch) ----
  wprep<<<1, 1024, 0, stream>>>(W1, WF1);
  wprep<<<1, 1024, 0, stream>>>(W2, WF2);

  // ---- conv1 ----
  hipMemsetAsync(buf1, 0, (size_t)N * 128 * sizeof(float), stream);
  hipMemsetAsync(cntN, 0, (size_t)N * sizeof(float), stream);
  gemm128_wmma<<<gemmGrd, 128, 0, stream>>>(x, WF1, b1, buf0, Mtiles);
  edge_pass<<<edgeBlk, 256, 0, stream>>>(buf0, src, dst, buf1, cntN, E);
  node_finish<<<(totNF + 255) / 256, 256, 0, stream>>>(buf1, cntN, buf0, totNF, 1);

  // ---- conv2 ----
  gemm128_wmma<<<gemmGrd, 128, 0, stream>>>(buf0, WF2, b2, buf1, Mtiles);
  hipMemsetAsync(buf0, 0, (size_t)N * 128 * sizeof(float), stream);
  hipMemsetAsync(cntN, 0, (size_t)N * sizeof(float), stream);
  edge_pass<<<edgeBlk, 256, 0, stream>>>(buf1, src, dst, buf0, cntN, E);

  // ---- global pool + head ----
  hipMemsetAsync(gacc, 0, (size_t)G * 128 * sizeof(float), stream);
  hipMemsetAsync(gcnt, 0, (size_t)G * sizeof(float), stream);
  node_pool<<<(totNF + 255) / 256, 256, 0, stream>>>(buf0, cntN, bat, gacc, gcnt, totNF);
  graph_head<<<G, 128, 0, stream>>>(gacc, gcnt, Wout, bout, (float*)d_out);
}
